// InstanceSegmentationHead_81578608820817
// MI455X (gfx1250) — matrix operand order repaired and checked by reading, not compiled, not run on hardware
//
#include <hip/hip_runtime.h>
#include <stdint.h>

typedef __attribute__((ext_vector_type(16))) _Float16 v16h;
typedef __attribute__((ext_vector_type(8)))  float    v8f;

#define NEGV  (-1e30f)
#define NTOT  6144     // 3*2048 boxes per batch
#define KKEEP 64

// ---------------------------------------------------------------------------
// Kernel 1: preprocess boxes -> scaled chw, xyxy, per-level masked scores
// ---------------------------------------------------------------------------
__global__ void prep_kernel(const float* __restrict__ b32, const float* __restrict__ b16,
                            const float* __restrict__ b8,
                            const float* __restrict__ s32, const float* __restrict__ s16,
                            const float* __restrict__ s8,
                            float* __restrict__ xyxy, float* __restrict__ chw,
                            float* __restrict__ msc) {
  int gid = blockIdx.x * 256 + threadIdx.x;
  if (gid >= 2 * NTOT) return;
  int b = gid / NTOT, i = gid - b * NTOT;
  int src = i >> 11, j = i & 2047;
  const float* bp = (src == 0 ? b32 : src == 1 ? b16 : b8) + ((size_t)b * 2048 + j) * 4;
  float cx = bp[0], cy = bp[1], w = bp[2], h = bp[3];
  float scale = (src == 0) ? 32.f : (src == 1) ? 16.f : 8.f;
  // level from UNscaled box
  float s = sqrtf(w * h);
  float lvl = floorf(3.0f + log2f(s / 224.0f));
  lvl = fminf(fmaxf(lvl, 1.0f), 4.0f);
  float scx = cx * scale, scy = cy * scale, sw = w * scale, sh = h * scale;
  size_t o4 = ((size_t)b * NTOT + i) * 4;
  chw[o4 + 0] = scx; chw[o4 + 1] = scy; chw[o4 + 2] = sw; chw[o4 + 3] = sh;
  xyxy[o4 + 0] = scx - 0.5f * sw; xyxy[o4 + 1] = scy - 0.5f * sh;
  xyxy[o4 + 2] = scx + 0.5f * sw; xyxy[o4 + 3] = scy + 0.5f * sh;
  float sc = (src == 0 ? s32 : src == 1 ? s16 : s8)[(size_t)b * 2048 + j];
#pragma unroll
  for (int t = 0; t < 3; ++t)
    msc[((size_t)(t * 2 + b)) * NTOT + i] = (lvl == (float)(t + 1)) ? sc : NEGV;
}

// ---------------------------------------------------------------------------
// Kernel 2: NMS. One 1024-thread block per (level,batch). Scores LDS-resident
// (staged via gfx1250 async-to-LDS), boxes register-resident, wave32 argmax.
// ---------------------------------------------------------------------------
__device__ __forceinline__ void red2(float& v, int& ix, float ov, int oi) {
  if (ov > v || (ov == v && oi < ix)) { v = ov; ix = oi; }
}

__global__ void nms_kernel(const float* __restrict__ xyxy, const float* __restrict__ msc,
                           int* __restrict__ sel) {
  __shared__ float sc_l[NTOT];
  __shared__ float rv[32];
  __shared__ int   ri[32];
  __shared__ float wb[4];
  __shared__ int   widx_l;
  __shared__ int   sel_l[KKEEP];

  int tid = threadIdx.x;
  int z = blockIdx.x, t = z >> 1, b = z & 1;
  const float* sbase = msc + (size_t)(t * 2 + b) * NTOT;
  unsigned long long sreg = (unsigned long long)(uintptr_t)sbase;

  // --- async stage the 24KB mutable score array into LDS (gfx1250 path) ---
#pragma unroll
  for (int k2 = 0; k2 < 6; ++k2) {
    int i = tid + (k2 << 10);
    unsigned ldsoff = (unsigned)(uintptr_t)(&sc_l[i]);
    unsigned goff = (unsigned)(i * 4);
    asm volatile("global_load_async_to_lds_b32 %0, %1, %2"
                 :: "v"(ldsoff), "v"(goff), "s"(sreg) : "memory");
  }
  // boxes (xyxy) -> registers, 6 per thread (L2 traffic, overlaps async DMA)
  float4 rb[6];
  const float4* xb = (const float4*)xyxy + (size_t)b * NTOT;
#pragma unroll
  for (int k2 = 0; k2 < 6; ++k2) rb[k2] = xb[tid + (k2 << 10)];

  asm volatile("s_wait_asynccnt 0" ::: "memory");
  __syncthreads();

  int lane = tid & 31, wv = tid >> 5;

  for (int it = 0; it < KKEEP; ++it) {
    // per-thread strided argmax (ascending scan => first-occurrence semantics)
    float best = -INFINITY; int bi = 0x7fffffff;
#pragma unroll
    for (int k2 = 0; k2 < 6; ++k2) {
      int i = tid + (k2 << 10);
      float s = sc_l[i];
      if (s > best) { best = s; bi = i; }
    }
    // wave32 butterfly reduce (value desc, index asc tie-break)
#pragma unroll
    for (int off = 16; off; off >>= 1) {
      float ov = __shfl_xor(best, off, 32);
      int   oi = __shfl_xor(bi, off, 32);
      red2(best, bi, ov, oi);
    }
    if (lane == 0) { rv[wv] = best; ri[wv] = bi; }
    __syncthreads();
    if (tid < 32) {
      float v = rv[tid]; int ix = ri[tid];
#pragma unroll
      for (int off = 16; off; off >>= 1) {
        float ov = __shfl_xor(v, off, 32);
        int   oi = __shfl_xor(ix, off, 32);
        red2(v, ix, ov, oi);
      }
      if (tid == 0) {
        widx_l = ix;
        sel_l[it] = (v > -5e29f) ? ix : -1;   // NEG/2 test
      }
    }
    __syncthreads();
    int j = widx_l;
    if (tid == (j & 1023)) {                   // owner broadcasts winner box
      float4 w4 = rb[j >> 10];
      wb[0] = w4.x; wb[1] = w4.y; wb[2] = w4.z; wb[3] = w4.w;
    }
    __syncthreads();
    // suppression (always applied, matching reference)
    float wx1 = wb[0], wy1 = wb[1], wx2 = wb[2], wy2 = wb[3];
    float wa = fmaxf(wx2 - wx1, 0.f) * fmaxf(wy2 - wy1, 0.f);
#pragma unroll
    for (int k2 = 0; k2 < 6; ++k2) {
      float4 bb = rb[k2];
      float ix1 = fmaxf(wx1, bb.x), iy1 = fmaxf(wy1, bb.y);
      float ix2 = fminf(wx2, bb.z), iy2 = fminf(wy2, bb.w);
      float inter = fmaxf(ix2 - ix1, 0.f) * fmaxf(iy2 - iy1, 0.f);
      float ba = fmaxf(bb.z - bb.x, 0.f) * fmaxf(bb.w - bb.y, 0.f);
      float iou = inter / (wa + ba - inter + 1e-9f);
      if (iou > 0.7f) sc_l[tid + (k2 << 10)] = NEGV;  // own entries only
    }
    // no barrier needed: next scan reads only this thread's own entries
  }
  __syncthreads();
  if (tid < KKEEP) sel[(size_t)(t * 2 + b) * KKEEP + tid] = sel_l[tid];
}

// ---------------------------------------------------------------------------
// Kernel 3: ROI-align. 1 block per (level,batch,box); 8 waves, 1 channel/wave.
// Bilinear gather + x-average in VALU; y-pool via v_wmma_f32_16x16x32_f16
// with constant pooling matrix A (exact 0.5 weights) — co-executes with VALU.
// ---------------------------------------------------------------------------
__global__ void roi_kernel(const float* __restrict__ p32, const float* __restrict__ p16,
                           const float* __restrict__ p8,
                           const float* __restrict__ chw, const int* __restrict__ sel,
                           float* __restrict__ out) {
  __shared__ float sxf[28], syf[28];
  __shared__ int   sx0[28], sx1[28], sy0[28], sy1[28];
  __shared__ float Sp[8][512];   // per-wave S' tile: 32 rows x 16 cols, zero-padded

  int tid = threadIdx.x, lane = tid & 31, wv = tid >> 5;
  int z = blockIdx.x;
  int t = z / 128, rem = z - t * 128, b = rem >> 6, k = rem & 63;
  int W = 28 << t, HW = W * W;
  const float* F = (t == 0 ? p32 : t == 1 ? p16 : p8) + (size_t)b * 256 * HW;

  int sv = sel[(size_t)(t * 2 + b) * KKEEP + k];
  float maskf = (sv >= 0) ? 1.f : 0.f;
  int idx = (sv < 0) ? 0 : sv;
  const float* bx = chw + ((size_t)b * NTOT + idx) * 4;
  // reference feeds scaled cxcywh straight into roi_align as x1,y1,x2,y2
  float bx1 = bx[0], by1 = bx[1], bx2 = bx[2], by2 = bx[3];
  float bwd = (bx2 - bx1) * (1.f / 14.f), bhd = (by2 - by1) * (1.f / 14.f);

  for (int s0 = tid; s0 < 8 * 512; s0 += 256) ((float*)Sp)[s0] = 0.f;

  if (tid < 28) {
    float tt = (float)(tid >> 1) + 0.25f + 0.5f * (float)(tid & 1);
    float c = fminf(fmaxf(bx1 + tt * bwd, 0.f), (float)(W - 1));
    float c0 = floorf(c);
    int i0 = (int)c0;
    sx0[tid] = i0; sxf[tid] = c - c0; sx1[tid] = (i0 + 1 < W - 1) ? i0 + 1 : W - 1;
  } else if (tid >= 32 && tid < 60) {
    int i = tid - 32;
    float tt = (float)(i >> 1) + 0.25f + 0.5f * (float)(i & 1);
    float c = fminf(fmaxf(by1 + tt * bhd, 0.f), (float)(W - 1));
    float c0 = floorf(c);
    int i0 = (int)c0;
    sy0[i] = i0; syf[i] = c - c0; sy1[i] = (i0 + 1 < W - 1) ? i0 + 1 : W - 1;
  }
  __syncthreads();

  // constant pooling matrix A (16x32 f16): A[m][kk] = 0.5 iff kk>>1 == m, m<14
  v16h amat;
#pragma unroll
  for (int e = 0; e < 16; ++e) {
    int v = e >> 1, h2 = e & 1;
    int kk = 2 * (v & 3) + h2 + ((v >= 4) ? 16 : 0) + ((lane >= 16) ? 8 : 0);
    int m = lane & 15;
    amat[e] = (_Float16)((((kk >> 1) == m) && (m < 14)) ? 0.5f : 0.0f);
  }

  int n = lane & 15;
  int koff = (lane >= 16) ? 8 : 0;
  size_t outBase = (((size_t)b * 192 + (size_t)t * 64 + k) * 256) * 196;

  for (int ci = 0; ci < 32; ++ci) {
    int c = ci * 8 + wv;
    const float* Fc = F + (size_t)c * HW;
    // build S'[i=0..27][ox=0..13]: bilinear samples, x-averaged in pairs
    for (int s = lane; s < 392; s += 32) {
      int i = s / 14, ox = s - i * 14;
      int y0 = sy0[i], y1 = sy1[i];
      float fy = syf[i];
      float acc2 = 0.f;
#pragma unroll
      for (int u = 0; u < 2; ++u) {
        int j = 2 * ox + u;
        int x0 = sx0[j], x1 = sx1[j];
        float fx = sxf[j];
        float v00 = Fc[y0 * W + x0], v01 = Fc[y0 * W + x1];
        float v10 = Fc[y1 * W + x0], v11 = Fc[y1 * W + x1];
        acc2 += (1.f - fy) * ((1.f - fx) * v00 + fx * v01)
              + fy * ((1.f - fx) * v10 + fx * v11);
      }
      Sp[wv][i * 16 + ox] = 0.5f * acc2;
    }
    asm volatile("" ::: "memory");
    __builtin_amdgcn_wave_barrier();   // per-wave LDS ordering (LDS is in-order)

    // B (32x16 f16): element e of v16h holds row K=(e&7)+16*(e>>3)(+8 for upper lanes), col n
    v16h bmat;
#pragma unroll
    for (int e = 0; e < 16; ++e)
      bmat[e] = (_Float16)Sp[wv][((e & 7) + koff + ((e >> 3) << 4)) * 16 + n];
    asm volatile("" ::: "memory");
    __builtin_amdgcn_wave_barrier();

    v8f acc = {};
    acc = __builtin_amdgcn_wmma_f32_16x16x32_f16(
        /*neg_a=*/false, amat, /*neg_b=*/false, bmat,
        /*c_mod=*/(short)0, acc, /*reuse_a=*/false, /*reuse_b=*/false);

    // D layout: VGPR r -> M=r (lanes 0-15) / M=r+8 (lanes 16-31), N=lane%16
#pragma unroll
    for (int r = 0; r < 8; ++r) {
      int M = r + ((lane >= 16) ? 8 : 0);
      if (M < 14 && n < 14)
        out[outBase + (size_t)c * 196 + M * 14 + n] = acc[r] * maskf;
    }
  }
}

// ---------------------------------------------------------------------------
extern "C" void kernel_launch(void* const* d_in, const int* in_sizes, int n_in,
                              void* d_out, int out_size, void* d_ws, size_t ws_size,
                              hipStream_t stream) {
  const float* p32 = (const float*)d_in[0];
  const float* p16 = (const float*)d_in[1];
  const float* p8  = (const float*)d_in[2];
  // d_in[3] = p4 (unused by the reference)
  const float* b32 = (const float*)d_in[4];
  const float* b16 = (const float*)d_in[5];
  const float* b8  = (const float*)d_in[6];
  const float* s32 = (const float*)d_in[7];
  const float* s16 = (const float*)d_in[8];
  const float* s8  = (const float*)d_in[9];

  float* ws   = (float*)d_ws;
  float* xyxy = ws;                       // 2*6144*4 = 49152 floats
  float* chw  = ws + 49152;               // 49152 floats
  float* msc  = ws + 98304;               // 3*2*6144 = 36864 floats
  int*   sel  = (int*)(ws + 135168);      // 3*2*64 = 384 ints

  prep_kernel<<<(2 * NTOT + 255) / 256, 256, 0, stream>>>(
      b32, b16, b8, s32, s16, s8, xyxy, chw, msc);
  nms_kernel<<<6, 1024, 0, stream>>>(xyxy, msc, sel);
  roi_kernel<<<384, 256, 0, stream>>>(p32, p16, p8, chw, sel, (float*)d_out);
}
